// GINE_35098472743461
// MI455X (gfx1250) — matrix operand order repaired and checked by reading, not compiled, not run on hardware
//
#include <hip/hip_runtime.h>
#include <hip/hip_bf16.h>

#define N_NODES 100000
#define N_EDGES 500000
#define N_GRAPH 2000
#define N_LAYER 5
#define EMB     300
#define H1      600
#define FEAT    256
#define BN_EPS  1e-5f

typedef __attribute__((ext_vector_type(16))) __bf16 v16bf;
typedef __attribute__((ext_vector_type(8)))  float  v8f;

// ---------------- elementwise / embedding kernels ----------------

__global__ void zero_f32(float* __restrict__ p, long n) {
    long i = (long)blockIdx.x * blockDim.x + threadIdx.x;
    if (i < n) p[i] = 0.0f;
}

__global__ void embed_kernel(const int* __restrict__ x,
                             const float* __restrict__ xe1,
                             const float* __restrict__ xe2,
                             float* __restrict__ h, long total) {
    long i = (long)blockIdx.x * blockDim.x + threadIdx.x;
    if (i >= total) return;
    long n = i / EMB;
    int  f = (int)(i - n * EMB);
    int t0 = x[n * 2 + 0];
    int t1 = x[n * 2 + 1];
    h[i] = xe1[(long)t0 * EMB + f] + xe2[(long)t1 * EMB + f];
}

__global__ void init_agg_kernel(const float* __restrict__ h,
                                const float* __restrict__ ee1l,
                                const float* __restrict__ ee2l,
                                float* __restrict__ agg, long total) {
    long i = (long)blockIdx.x * blockDim.x + threadIdx.x;
    if (i >= total) return;
    int f = (int)(i % EMB);
    // self-loop edge has attr (4, 0)
    agg[i] = h[i] + ee1l[4 * EMB + f] + ee2l[0 * EMB + f];
}

// one block per edge; 300 active lanes; atomic scatter-add at dst
__global__ void edge_scatter_kernel(const float* __restrict__ h,
                                    const int* __restrict__ ei,
                                    const int* __restrict__ ea,
                                    const float* __restrict__ ee1l,
                                    const float* __restrict__ ee2l,
                                    float* __restrict__ agg) {
    int e = blockIdx.x;
    int f = threadIdx.x;
    if (f >= EMB) return;
    int s  = ei[e];             // edge_index[0][e]
    int d  = ei[N_EDGES + e];   // edge_index[1][e]
    int a0 = ea[e * 2 + 0];
    int a1 = ea[e * 2 + 1];
    float msg = h[(long)s * EMB + f] + ee1l[a0 * EMB + f] + ee2l[a1 * EMB + f];
    atomicAdd(&agg[(long)d * EMB + f], msg);
}

// ---------------- weight packing into WMMA B-fragment layout ----------------
// Wp[((kt*NT + nt)*32 + lane)*16 + i] = bf16(W[kk, col]), zero-padded at K/Nc tails
//   col = nt*16 + (lane&15),  kk = kt*32 + (i<8 ? (lane>>4)*8 + i : 16 + (lane>>4)*8 + (i-8))

__global__ void pack_w_kernel(const float* __restrict__ W, __bf16* __restrict__ Wp,
                              int K, int Nc, int NT, long total) {
    long idx = (long)blockIdx.x * blockDim.x + threadIdx.x;
    if (idx >= total) return;
    int  i    = (int)(idx & 15);
    int  lane = (int)((idx >> 4) & 31);
    long t    = idx >> 9;            // kt*NT + nt
    int  nt   = (int)(t % NT);
    int  kt   = (int)(t / NT);
    int  col  = nt * 16 + (lane & 15);
    int  kk   = kt * 32 + ((i < 8) ? ((lane >> 4) * 8 + i) : (16 + (lane >> 4) * 8 + (i - 8)));
    float v = 0.0f;
    if (col < Nc && kk < K) v = W[(long)kk * Nc + col];
    Wp[idx] = (__bf16)v;
}

// ---------------- WMMA GEMM: C[M,Nc] = act(A[M,K] @ W[K,Nc] + bias) ----------------

__device__ inline v16bf load_a_full(const float* __restrict__ Ar, int k0, int half) {
    // two contiguous 8-float runs per lane -> 4x b128 loads (16B aligned)
    const float4* p = (const float4*)(Ar + k0 + half * 8);
    const float4* q = (const float4*)(Ar + k0 + 16 + half * 8);
    float4 f0 = p[0], f1 = p[1], f2 = q[0], f3 = q[1];
    v16bf a;
    a[0]  = (__bf16)f0.x; a[1]  = (__bf16)f0.y; a[2]  = (__bf16)f0.z; a[3]  = (__bf16)f0.w;
    a[4]  = (__bf16)f1.x; a[5]  = (__bf16)f1.y; a[6]  = (__bf16)f1.z; a[7]  = (__bf16)f1.w;
    a[8]  = (__bf16)f2.x; a[9]  = (__bf16)f2.y; a[10] = (__bf16)f2.z; a[11] = (__bf16)f2.w;
    a[12] = (__bf16)f3.x; a[13] = (__bf16)f3.y; a[14] = (__bf16)f3.z; a[15] = (__bf16)f3.w;
    return a;
}

__device__ inline v16bf load_a_tail(const float* __restrict__ Ar, int K, int k0, int half) {
    // addresses clamped (always legal), value zeroed by select -> no exec branches
    v16bf a;
#pragma unroll
    for (int i = 0; i < 16; ++i) {
        int kk = (i < 8) ? (k0 + half * 8 + i) : (k0 + 16 + half * 8 + (i - 8));
        int kc = (kk < K) ? kk : (K - 1);
        float x = Ar[kc];
        a[i] = (__bf16)((kk < K) ? x : 0.0f);
    }
    return a;
}

// block = 128 threads = 4 waves; wave -> one 16-row tile x four 16-col tiles
__global__ void gemm_bias_act_kernel(const float* __restrict__ A,
                                     const v16bf* __restrict__ Wp,
                                     const float* __restrict__ bias,
                                     float* __restrict__ C,
                                     int M, int K, int Nc, int NT, int do_relu) {
    int wave = threadIdx.x >> 5;
    int lane = threadIdx.x & 31;
    int half = lane >> 4;
    int ln   = lane & 15;
    int mt = blockIdx.x * 4 + wave;
    if (mt * 16 >= M) return;                 // wave-uniform: EXEC stays all-ones
    int row = mt * 16 + ln;
    int ntb = blockIdx.y * 4;
    int nt[4];
#pragma unroll
    for (int t = 0; t < 4; ++t) {
        int v = ntb + t;
        nt[t] = (v < NT) ? v : (NT - 1);      // duplicate compute, store guarded
    }

    const float* Ar = A + (long)row * K;
    v8f c[4];
#pragma unroll
    for (int t = 0; t < 4; ++t) c[t] = (v8f){};

    int KFULL = K & ~31;
    int kt = 0;
    for (int k0 = 0; k0 < KFULL; k0 += 32, ++kt) {
        __builtin_prefetch(Ar + k0 + 32, 0, 1);
        v16bf a = load_a_full(Ar, k0, half);
        const v16bf* wrow = Wp + ((long)kt * NT) * 32 + lane;
        v16bf b0 = wrow[nt[0] * 32];
        v16bf b1 = wrow[nt[1] * 32];
        v16bf b2 = wrow[nt[2] * 32];
        v16bf b3 = wrow[nt[3] * 32];
        c[0] = __builtin_amdgcn_wmma_f32_16x16x32_bf16(false, a, false, b0,
                                                       (short)0, c[0], false, false);
        c[1] = __builtin_amdgcn_wmma_f32_16x16x32_bf16(false, a, false, b1,
                                                       (short)0, c[1], false, false);
        c[2] = __builtin_amdgcn_wmma_f32_16x16x32_bf16(false, a, false, b2,
                                                       (short)0, c[2], false, false);
        c[3] = __builtin_amdgcn_wmma_f32_16x16x32_bf16(false, a, false, b3,
                                                       (short)0, c[3], false, false);
    }
    if (KFULL < K) {
        v16bf a = load_a_tail(Ar, K, KFULL, half);
        const v16bf* wrow = Wp + ((long)kt * NT) * 32 + lane;
        v16bf b0 = wrow[nt[0] * 32];
        v16bf b1 = wrow[nt[1] * 32];
        v16bf b2 = wrow[nt[2] * 32];
        v16bf b3 = wrow[nt[3] * 32];
        c[0] = __builtin_amdgcn_wmma_f32_16x16x32_bf16(false, a, false, b0,
                                                       (short)0, c[0], false, false);
        c[1] = __builtin_amdgcn_wmma_f32_16x16x32_bf16(false, a, false, b1,
                                                       (short)0, c[1], false, false);
        c[2] = __builtin_amdgcn_wmma_f32_16x16x32_bf16(false, a, false, b2,
                                                       (short)0, c[2], false, false);
        c[3] = __builtin_amdgcn_wmma_f32_16x16x32_bf16(false, a, false, b3,
                                                       (short)0, c[3], false, false);
    }

#pragma unroll
    for (int t = 0; t < 4; ++t) {
        int col = (ntb + t) * 16 + ln;        // unclamped for guarding
        if (col < Nc) {
            float bv = bias[col];
#pragma unroll
            for (int r = 0; r < 8; ++r) {
                int orow = mt * 16 + r + half * 8;
                float v = c[t][r] + bv;
                if (do_relu) v = fmaxf(v, 0.0f);
                C[(long)orow * Nc + col] = v;
            }
        }
    }
}

// ---------------- BatchNorm (two pass) ----------------

#define BN_ROWS 128
__global__ void bn_stats_kernel(const float* __restrict__ h,
                                float* __restrict__ stats) {
    __shared__ float s[2 * EMB];
    int tid = threadIdx.x;
    for (int i = tid; i < 2 * EMB; i += blockDim.x) s[i] = 0.0f;
    __syncthreads();
    int r0 = blockIdx.x * BN_ROWS;
    int rows = min(BN_ROWS, N_NODES - r0);
    int items = rows * EMB;
    for (int idx = tid; idx < items; idx += blockDim.x) {
        int r = idx / EMB;
        int f = idx - r * EMB;
        float v = h[(long)(r0 + r) * EMB + f];
        atomicAdd(&s[f], v);
        atomicAdd(&s[EMB + f], v * v);
    }
    __syncthreads();
    for (int f = tid; f < EMB; f += blockDim.x) {
        atomicAdd(&stats[f], s[f]);
        atomicAdd(&stats[EMB + f], s[EMB + f]);
    }
}

__global__ void bn_apply_kernel(float* __restrict__ h,
                                const float* __restrict__ stats,
                                const float* __restrict__ gamma,
                                const float* __restrict__ beta,
                                long total, float invN) {
    long i = (long)blockIdx.x * blockDim.x + threadIdx.x;
    if (i >= total) return;
    int f = (int)(i % EMB);
    float mu  = stats[f] * invN;
    float var = stats[EMB + f] * invN - mu * mu;
    float v = (h[i] - mu) * rsqrtf(var + BN_EPS) * gamma[f] + beta[f];
    h[i] = fmaxf(v, 0.0f);
}

// ---------------- pooling + head ----------------

__global__ void pool_scatter_kernel(const float* __restrict__ hf,
                                    const int* __restrict__ batch,
                                    float* __restrict__ gsum,
                                    float* __restrict__ gcnt) {
    int n = blockIdx.x;
    int f = threadIdx.x;
    int g = batch[n];
    atomicAdd(&gsum[(long)g * FEAT + f], hf[(long)n * FEAT + f]);
    if (f == 0) atomicAdd(&gcnt[g], 1.0f);
}

__global__ void pool_fin_kernel(const float* __restrict__ gsum,
                                const float* __restrict__ gcnt,
                                float* __restrict__ pooled, int total) {
    int i = blockIdx.x * blockDim.x + threadIdx.x;
    if (i >= total) return;
    int g = i >> 8;  // FEAT == 256
    pooled[i] = gsum[i] / fmaxf(gcnt[g], 1.0f);
}

__global__ void head_kernel(const float* __restrict__ pooled,
                            const float* __restrict__ pw1,
                            const float* __restrict__ pb1,
                            const float* __restrict__ pw2,
                            const float* __restrict__ pb2,
                            float* __restrict__ pred) {
    __shared__ float hid[FEAT / 2];
    int g = blockIdx.x;
    int j = threadIdx.x;  // 0..127
    const float* p = pooled + (long)g * FEAT;
    float s = pb1[j];
    for (int k = 0; k < FEAT; ++k) s += p[k] * pw1[k * (FEAT / 2) + j];
    hid[j] = fmaxf(s, 0.0f);
    __syncthreads();
    if (j < 2) {
        float o = pb2[j];
        for (int k = 0; k < FEAT / 2; ++k) o += hid[k] * pw2[k * 2 + j];
        pred[g * 2 + j] = o;
    }
}

// ---------------- launcher ----------------

extern "C" void kernel_launch(void* const* d_in, const int* in_sizes, int n_in,
                              void* d_out, int out_size, void* d_ws, size_t ws_size,
                              hipStream_t stream) {
    const int*   x     = (const int*)d_in[0];
    const int*   ei    = (const int*)d_in[1];
    const int*   ea    = (const int*)d_in[2];
    const int*   batch = (const int*)d_in[3];
    const float* xe1   = (const float*)d_in[4];
    const float* xe2   = (const float*)d_in[5];
    const float* ee1   = (const float*)d_in[6];
    const float* ee2   = (const float*)d_in[7];
    const float* w1    = (const float*)d_in[8];
    const float* b1    = (const float*)d_in[9];
    const float* w2    = (const float*)d_in[10];
    const float* b2    = (const float*)d_in[11];
    const float* gamma = (const float*)d_in[12];
    const float* beta  = (const float*)d_in[13];
    const float* fw    = (const float*)d_in[14];
    const float* fb    = (const float*)d_in[15];
    const float* pw1   = (const float*)d_in[16];
    const float* pb1   = (const float*)d_in[17];
    const float* pw2   = (const float*)d_in[18];
    const float* pb2   = (const float*)d_in[19];

    char* ws = (char*)d_ws;
    float*  h      = (float*)(ws + 0UL);            // N*EMB*4   = 120,000,000 B
    float*  agg    = (float*)(ws + 120000000UL);    // N*EMB*4   = 120,000,000 B
    float*  hidden = (float*)(ws + 240000000UL);    // N*H1*4    = 240,000,000 B (reused as hfeat)
    float*  stats  = (float*)(ws + 480000000UL);    // 600 f32
    float*  gsum   = (float*)(ws + 480001024UL);    // G*FEAT*4  = 2,048,000 B
    float*  gcnt   = (float*)(ws + 482049024UL);    // G*4
    __bf16* wp1    = (__bf16*)(ws + 482057216UL);   // 10*38*512 bf16 = 389,120 B
    __bf16* wp2    = (__bf16*)(ws + 482446336UL);   // 19*19*512 bf16 = 369,664 B
    __bf16* wpf    = (__bf16*)(ws + 482816000UL);   // 10*16*512 bf16 = 163,840 B

    const long totEmb  = (long)N_NODES * EMB;       // 30M
    const int  gridEmb = (int)((totEmb + 255) / 256);
    const int  mblocks = (N_NODES / 16 + 3) / 4;    // 6250 M-tiles -> 1563 blocks

    // fragment-tile counts (K tiles of 32, col tiles of 16), zero-padded in pack
    const int NT1 = (H1 + 15) / 16;                 // 38
    const int NT2 = (EMB + 15) / 16;                // 19
    const int NTF = FEAT / 16;                      // 16
    const long tp1 = (long)((EMB + 31) / 32) * NT1 * 512;  // 194,560
    const long tp2 = (long)((H1 + 31) / 32) * NT2 * 512;   // 184,832
    const long tpf = (long)((EMB + 31) / 32) * NTF * 512;  //  81,920

    // h = xemb1[x0] + xemb2[x1]
    embed_kernel<<<gridEmb, 256, 0, stream>>>(x, xe1, xe2, h, totEmb);
    // feature-projection weights packed once
    pack_w_kernel<<<(int)((tpf + 255) / 256), 256, 0, stream>>>(fw, wpf, EMB, FEAT, NTF, tpf);

    for (int l = 0; l < N_LAYER; ++l) {
        const float* ee1l = ee1 + (long)l * 6 * EMB;
        const float* ee2l = ee2 + (long)l * 3 * EMB;
        const float* w1l  = w1 + (long)l * EMB * H1;
        const float* b1l  = b1 + (long)l * H1;
        const float* w2l  = w2 + (long)l * H1 * EMB;
        const float* b2l  = b2 + (long)l * EMB;
        const float* gl   = gamma + (long)l * EMB;
        const float* bl   = beta + (long)l * EMB;

        // pack this layer's weights into WMMA fragment layout (tiny)
        pack_w_kernel<<<(int)((tp1 + 255) / 256), 256, 0, stream>>>(w1l, wp1, EMB, H1, NT1, tp1);
        pack_w_kernel<<<(int)((tp2 + 255) / 256), 256, 0, stream>>>(w2l, wp2, H1, EMB, NT2, tp2);

        // agg = h + self_emb ; then scatter-add messages
        init_agg_kernel<<<gridEmb, 256, 0, stream>>>(h, ee1l, ee2l, agg, totEmb);
        edge_scatter_kernel<<<N_EDGES, 320, 0, stream>>>(h, ei, ea, ee1l, ee2l, agg);

        // hidden = relu(agg @ w1 + b1)   [N,300]x[300,600]
        gemm_bias_act_kernel<<<dim3(mblocks, (NT1 + 3) / 4), 128, 0, stream>>>(
            agg, (const v16bf*)wp1, b1l, hidden, N_NODES, EMB, H1, NT1, 1);
        // h = hidden @ w2 + b2           [N,600]x[600,300]
        gemm_bias_act_kernel<<<dim3(mblocks, (NT2 + 3) / 4), 128, 0, stream>>>(
            hidden, (const v16bf*)wp2, b2l, h, N_NODES, H1, EMB, NT2, 0);

        // BatchNorm (training-mode batch stats) + ReLU
        zero_f32<<<3, 256, 0, stream>>>(stats, 2 * EMB);
        bn_stats_kernel<<<(N_NODES + BN_ROWS - 1) / BN_ROWS, 256, 0, stream>>>(h, stats);
        bn_apply_kernel<<<gridEmb, 256, 0, stream>>>(h, stats, gl, bl, totEmb,
                                                     1.0f / (float)N_NODES);
    }

    // hfeat = h @ feat_w + feat_b   [N,300]x[300,256]  (reuse hidden buffer)
    float* hfeat = hidden;
    gemm_bias_act_kernel<<<dim3(mblocks, (NTF + 3) / 4), 128, 0, stream>>>(
        h, (const v16bf*)wpf, fb, hfeat, N_NODES, EMB, FEAT, NTF, 0);

    // global mean pool
    const long gsz = (long)N_GRAPH * FEAT;
    zero_f32<<<(int)((gsz + 255) / 256), 256, 0, stream>>>(gsum, gsz);
    zero_f32<<<(N_GRAPH + 255) / 256, 256, 0, stream>>>(gcnt, N_GRAPH);
    pool_scatter_kernel<<<N_NODES, FEAT, 0, stream>>>(hfeat, batch, gsum, gcnt);

    float* pooled = (float*)d_out;                    // [G, FEAT]
    float* pred   = pooled + (long)N_GRAPH * FEAT;    // [G, 2]
    pool_fin_kernel<<<(int)((gsz + 255) / 256), 256, 0, stream>>>(gsum, gcnt, pooled,
                                                                  (int)gsz);
    head_kernel<<<N_GRAPH, FEAT / 2, 0, stream>>>(pooled, pw1, pb1, pw2, pb2, pred);
}
